// LocalDeformKernel2d_89507118448686
// MI455X (gfx1250) — matrix operand order repaired and checked by reading, not compiled, not run on hardware
//
#include <hip/hip_runtime.h>

// ---------------------------------------------------------------------------
// LocalDeformKernel2d on MI455X (gfx1250, wave32, WMMA)
//   B=4, C=128, H=W=96, K=3, KS=4, PAD=1, rot channels O=18
// Phase 0a: pack rot_w into WMMA B-fragment layout (f16, N padded to 32)
// Phase 0b: pack x into zero-padded channel-last fragment-ordered f16 tensor
// Phase 1:  rot = conv3x3(x, rot_w) as implicit GEMM, v_wmma_f32_16x16x32_f16
//           (A fragment = ONE aligned 32B load per lane per K-step)
// Phase 2:  bilinear-sample weight(128,4,4) at rot coords, dot with patches
// ---------------------------------------------------------------------------

typedef __attribute__((ext_vector_type(16))) _Float16 v16h;
typedef __attribute__((ext_vector_type(8)))  float    v8f;

#define B_   4
#define C_   128
#define H_   96
#define W_   96
#define HW_  (H_ * W_)
#define HP_  (H_ + 2)     // padded
#define WP_  (W_ + 2)
#define O_   18           // rot output channels (padded to 32 = two N tiles)
#define KTOT (C_ * 9)     // GEMM reduction length = 1152
#define NKB  (KTOT / 32)  // 36 K-steps of 32

#define BPACK_BYTES (2 * NKB * 32 * 16 * 2)            // 73728, 256-aligned
#define XH_ELEMS    (B_ * HP_ * WP_ * C_)              // 4,917,248 halves

// Swap bits 3 and 4 of a channel index: maps channel -> stored position so
// that each lane-half's 16 fragment values {koff+0..7, 16+koff+0..7} are
// contiguous.  Involution, so it is its own inverse.
__device__ __forceinline__ int swap34(int r) {
    return (r & ~0x18) | ((r & 8) << 1) | ((r & 16) >> 1);
}

// ---------------------------------------------------------------------------
// Pack rot_w (18,128,3,3) -> bpack[nt][kb][lane][i] (f16), zero-padded to 32
// output channels.  B (32x16, f16) lane layout: n = lane&15,
// K(lane,i) = swap34((lane>>4)*16 + i).  k = ij*128 + c -> rot_w[n*1152 + c*9 + ij].
// ---------------------------------------------------------------------------
__global__ __launch_bounds__(256) void pack_b_kernel(
    const float* __restrict__ rot_w, _Float16* __restrict__ bpack)
{
    int tid = blockIdx.x * 256 + threadIdx.x;   // == ((nt*NKB+kb)*32+lane)*16+i
    int i    = tid & 15;
    int lane = (tid >> 4) & 31;
    int r    = tid >> 9;
    int kb   = r % NKB;
    int nt   = r / NKB;
    if (nt >= 2) return;

    int n = nt * 16 + (lane & 15);
    int k = kb * 32 + swap34((lane >> 4) * 16 + i);
    int ij = k >> 7;
    int c  = k & 127;
    float v = (n < O_) ? rot_w[n * KTOT + c * 9 + ij] : 0.0f;
    bpack[tid] = (_Float16)v;
}

// ---------------------------------------------------------------------------
// Pack x (4,128,96,96) f32 -> xh[b][hp][wp][pos] f16, zero-padded spatially,
// channels permuted by swap34 so A fragments are contiguous.
// ---------------------------------------------------------------------------
__global__ __launch_bounds__(256) void pack_x_kernel(
    const float* __restrict__ x, _Float16* __restrict__ xh)
{
    int tid = blockIdx.x * 256 + threadIdx.x;
    if (tid >= XH_ELEMS) return;
    int p  = tid & 127;
    int s  = tid >> 7;
    int wp = s % WP_;
    s      = s / WP_;
    int hp = s % HP_;
    int b  = s / HP_;
    int r  = swap34(p);          // stored position p holds channel r
    float v = 0.0f;
    if (hp >= 1 && hp <= H_ && wp >= 1 && wp <= W_)
        v = x[((b * C_ + r) * H_ + (hp - 1)) * W_ + (wp - 1)];
    xh[tid] = (_Float16)v;
}

// ---------------------------------------------------------------------------
// Fused main kernel: 1 wave = 16 consecutive pixels of one row; block = 4
// waves.  (16 | W and 16 | H*W, so b and h are uniform within a tile.)
// ---------------------------------------------------------------------------
__global__ __launch_bounds__(128) void deform_main_kernel(
    const float* __restrict__ x,
    const float* __restrict__ rot_b,
    const float* __restrict__ weight,       // (128,4,4) = (128,16)
    const _Float16* __restrict__ bpack,
    const _Float16* __restrict__ xh,
    float* __restrict__ out)
{
    __shared__ float    wlds[C_ * 16];        // 8 KB weight table
    __shared__ float    rot_lds[4][16 * 32];  // per-wave rot transpose
    __shared__ float4   coef_lds[4][144];     // per-(pixel,tap) bilinear coeffs
    __shared__ unsigned idx_lds[4][144];      // packed 4x corner indices

    const int t    = threadIdx.x;
    const int wave = t >> 5;
    const int lane = t & 31;
    const int m    = lane & 15;               // pixel within tile

    // Stage the 4x4-per-channel weight table into LDS (read in phase 2b).
    for (int s = t; s < C_ * 16; s += 128) wlds[s] = weight[s];

    // Tile coordinates.
    const int tile = blockIdx.x * 4 + wave;
    const int p0   = tile * 16;
    const int b    = p0 / HW_;
    const int rem  = p0 - b * HW_;
    const int h    = rem / W_;
    const int w0   = rem - h * W_;
    const int w    = w0 + m;

    // Per-lane tap addresses: padded f16 tensor (phase 1, no masks needed)
    // and clamped f32 addresses + masks (phase 2b, full precision).
    int   pixoff[9];      // element offset (halves) into xh, channel 0
    int   xoff[9];        // element offset into f32 x (clamped)
    float msk[9];
#pragma unroll
    for (int ij = 0; ij < 9; ++ij) {
        int di = ij / 3 - 1, dj = ij % 3 - 1;
        pixoff[ij] = ((b * HP_ + (h + di + 1)) * WP_ + (w + dj + 1)) * C_;
        int hh = h + di, ww = w + dj;
        bool ok = (hh >= 0) & (hh < H_) & (ww >= 0) & (ww < W_);
        int hc = min(max(hh, 0), H_ - 1);
        int wc = min(max(ww, 0), W_ - 1);
        xoff[ij] = b * C_ * HW_ + hc * W_ + wc;
        msk[ij]  = ok ? 1.0f : 0.0f;
    }

    // ---------------- Phase 1: implicit-GEMM conv via WMMA -----------------
    // A fragment: one aligned 32-byte v16h load per lane per K-step.
    const int halfsel = (lane >> 4) << 4;     // 0 or 16 halves
    const v16h* bp = (const v16h*)bpack;

    v8f acc0 = {};   // rot channels 0..15
    v8f acc1 = {};   // rot channels 16..17 (+ zero pad)
#pragma unroll
    for (int ij = 0; ij < 9; ++ij) {
        const v16h* ap = (const v16h*)(xh + pixoff[ij] + halfsel);
#pragma unroll
        for (int cb = 0; cb < 4; ++cb) {      // 32 channels per K-step
            v16h a  = ap[cb * 2];             // cb*32 halves
            int  kb = ij * 4 + cb;
            v16h b0 = bp[kb * 32 + lane];
            v16h b1 = bp[(NKB + kb) * 32 + lane];
            acc0 = __builtin_amdgcn_wmma_f32_16x16x32_f16(
                       false, a, false, b0, (short)0, acc0, false, false);
            acc1 = __builtin_amdgcn_wmma_f32_16x16x32_f16(
                       false, a, false, b1, (short)0, acc1, false, false);
        }
    }

    // Transpose accumulators through LDS: lane holds column n = lane&15,
    // rows M = (lane>>4)*8 + i.
    {
        const int mrow = (lane >> 4) * 8;
        const int n    = lane & 15;
#pragma unroll
        for (int i = 0; i < 8; ++i) {
            rot_lds[wave][(mrow + i) * 32 + n]      = acc0[i];
            rot_lds[wave][(mrow + i) * 32 + n + 16] = acc1[i];
        }
    }
    __syncthreads();

    // -------- Phase 2a: bilinear coefficients per (pixel, tap) -------------
#pragma unroll
    for (int it = 0; it < 5; ++it) {
        int task = lane + it * 32;                 // task = ij*16 + pixel
        if (task < 144) {
            int tm = task & 15, ij = task >> 4;
            int di = ij / 3, dj = ij % 3;
            float ch = rot_lds[wave][tm * 32 + 2 * ij]     + rot_b[2 * ij]
                     + (0.5f + (float)di);
            float cw = rot_lds[wave][tm * 32 + 2 * ij + 1] + rot_b[2 * ij + 1]
                     + (0.5f + (float)dj);
            ch = fminf(fmaxf(ch, 0.0f), 3.0f);
            cw = fminf(fmaxf(cw, 0.0f), 3.0f);
            float fh = floorf(ch), fw = floorf(cw);
            int h0 = (int)fh, w0i = (int)fw;
            int h1 = min(h0 + 1, 3), w1 = min(w0i + 1, 3);
            float lh = ch - fh, lw = cw - fw;
            coef_lds[wave][task] = make_float4((1.0f - lh) * (1.0f - lw),
                                               (1.0f - lh) * lw,
                                               lh * (1.0f - lw),
                                               lh * lw);
            idx_lds[wave][task] =
                (unsigned)(h0 * 4 + w0i)        |
                ((unsigned)(h0 * 4 + w1)  << 8) |
                ((unsigned)(h1 * 4 + w0i) << 16)|
                ((unsigned)(h1 * 4 + w1)  << 24);
        }
    }
    __syncthreads();

    // -------- Phase 2b: channel loop, LDS weight gather + patch FMA --------
    float4   cf[9];
    unsigned idp[9];
#pragma unroll
    for (int ij = 0; ij < 9; ++ij) {
        cf[ij]  = coef_lds[wave][ij * 16 + m];
        idp[ij] = idx_lds[wave][ij * 16 + m];
    }

    const int chalf = lane >> 4;                   // 2 channels in flight per m
    const int obase = b * C_ * HW_ + h * W_ + w;
    for (int cp = 0; cp < 64; ++cp) {
        int c = cp * 2 + chalf;
        // Stream the next channels' center patch line into cache.
        __builtin_prefetch(x + xoff[4] + min(c + 8, C_ - 1) * HW_, 0, 1);
        const float* wr = wlds + c * 16;
        float sum = 0.0f;
#pragma unroll
        for (int ij = 0; ij < 9; ++ij) {
            unsigned id = idp[ij];
            float ws = cf[ij].x * wr[id & 15]
                     + cf[ij].y * wr[(id >> 8)  & 15]
                     + cf[ij].z * wr[(id >> 16) & 15]
                     + cf[ij].w * wr[(id >> 24) & 15];
            float xv = x[xoff[ij] + c * HW_] * msk[ij];
            sum += ws * xv;
        }
        out[obase + c * HW_] = sum;
    }
}

// ---------------------------------------------------------------------------
extern "C" void kernel_launch(void* const* d_in, const int* in_sizes, int n_in,
                              void* d_out, int out_size, void* d_ws, size_t ws_size,
                              hipStream_t stream)
{
    const float* x      = (const float*)d_in[0];
    const float* rot_w  = (const float*)d_in[1];
    const float* rot_b  = (const float*)d_in[2];
    const float* weight = (const float*)d_in[3];
    float*       out    = (float*)d_out;

    _Float16* bpack = (_Float16*)d_ws;                          // 73,728 B
    _Float16* xh    = (_Float16*)((char*)d_ws + BPACK_BYTES);   // 9,834,496 B

    // Phase 0a: B fragments (2 * 36 * 512 = 36,864 elements).
    pack_b_kernel<<<144, 256, 0, stream>>>(rot_w, bpack);

    // Phase 0b: padded channel-last f16 x (4,917,248 elements).
    pack_x_kernel<<<(XH_ELEMS + 255) / 256, 256, 0, stream>>>(x, xh);

    // Phase 1+2: 36,864 pixels / 16 per wave = 2304 tiles; grid is 2304 blocks x 4 waves.
    deform_main_kernel<<<2304, 128, 0, stream>>>(x, rot_b, weight, bpack, xh, out);
}
